// CNFEncoder_14139032338992
// MI455X (gfx1250) — matrix operand order repaired and checked by reading, not compiled, not run on hardware
//
#include <hip/hip_runtime.h>
#include <hip/hip_bf16.h>

typedef __attribute__((ext_vector_type(16))) __bf16 v16bf;
typedef __attribute__((ext_vector_type(8)))  float  v8f;

union ABfrag {
    v16bf        v;
    unsigned int u[8];
    uint4        q[2];
};

// pack two f32 -> one VGPR holding (bf16(lo) in [15:0], bf16(hi) in [31:16])
__device__ __forceinline__ unsigned int pk_bf16(float lo, float hi) {
    unsigned short a = __builtin_bit_cast(unsigned short, (__bf16)lo);
    unsigned short b = __builtin_bit_cast(unsigned short, (__bf16)hi);
    return (unsigned int)a | ((unsigned int)b << 16);
}

// ---------------------------------------------------------------------------
// WMMA GEMM:  Y[M x 128] = X[M x KTOT] @ W[KTOT x 128] + bias
//   KTOT = K0 + (EXTRA ? 1 : 0); X(row,k) = k < K0 ? X0[row*K0+k] : X1[row]
// One wave owns a 16-row x 128-col stripe (8 f32 accumulators).
// Weight slab (32 K-rows) is bf16-packed into LDS per step in [col][pair]
// layout (stride 20 words) so each B fragment is two ds_load_b128 that land
// in the contiguous even-aligned VGPR octet WMMA needs. All K bounds are
// compile-time -> unconditional loads everywhere.
// ---------------------------------------------------------------------------
template <int K0, bool EXTRA>
__global__ __launch_bounds__(128)
void gemm_wmma_bias(const float* __restrict__ X0, const float* __restrict__ X1,
                    const float* __restrict__ W,  const float* __restrict__ bias,
                    float* __restrict__ Y, int M) {
    constexpr int KTOT  = K0 + (EXTRA ? 1 : 0);
    constexpr int KPAD  = (KTOT + 31) & ~31;
    constexpr int KFULL = (KTOT / 32) * 32;      // whole 32-K slabs
    constexpr int CSTR  = 20;                    // LDS column stride (words)

    __shared__ unsigned int ldsb[128 * CSTR];    // one 32-K weight slab, bf16 pairs

    const int tid  = threadIdx.x;
    const int wave = tid >> 5;
    const int lane = tid & 31;
    const int m    = lane & 15;
    const int half = lane >> 4;
    const int rowBase = (blockIdx.x * 4 + wave) * 16;
    const int row  = rowBase + m;
    const int srow = (row < M) ? row : (M - 1);  // clamp: loads always in-bounds

    v8f acc[8] = {};

    // ---------------- full 32-K slabs (no guards anywhere) ----------------
    for (int k0 = 0; k0 < KFULL; k0 += 32) {
        __syncthreads();
#pragma unroll
        for (int p = 0; p < 16; ++p) {           // stage W slab, coalesced rows
            float w0 = W[(size_t)(k0 + 2 * p)     * 128 + tid];
            float w1 = W[(size_t)(k0 + 2 * p + 1) * 128 + tid];
            ldsb[tid * CSTR + p] = pk_bf16(w0, w1);
        }
        __syncthreads();

        // A fragment: lane holds row srow, K = k0+16h..+15
        const float* xrow = X0 + (size_t)srow * K0 + k0 + 16 * half;
        float4 g0 = *(const float4*)(xrow);
        float4 g1 = *(const float4*)(xrow + 4);
        float4 g2 = *(const float4*)(xrow + 8);
        float4 g3 = *(const float4*)(xrow + 12);
        ABfrag a;
        a.u[0] = pk_bf16(g0.x, g0.y); a.u[1] = pk_bf16(g0.z, g0.w);
        a.u[2] = pk_bf16(g1.x, g1.y); a.u[3] = pk_bf16(g1.z, g1.w);
        a.u[4] = pk_bf16(g2.x, g2.y); a.u[5] = pk_bf16(g2.z, g2.w);
        a.u[6] = pk_bf16(g3.x, g3.y); a.u[7] = pk_bf16(g3.z, g3.w);

#pragma unroll
        for (int nt = 0; nt < 8; ++nt) {
            const int col = nt * 16 + m;
            const unsigned int* bp = &ldsb[col * CSTR + 8 * half];
            ABfrag b;
            b.q[0] = *(const uint4*)(bp);        // pairs q .. q+3
            b.q[1] = *(const uint4*)(bp + 4);    // pairs q+4 .. q+7
            acc[nt] = __builtin_amdgcn_wmma_f32_16x16x32_bf16(
                false, a.v, false, b.v, (short)0, acc[nt], false, false);
        }
    }

    // ---------------- tail slab (K0==8 partial, or EXTRA column) ----------
    if constexpr (KPAD > KFULL) {
        __syncthreads();
#pragma unroll
        for (int p = 0; p < 16; ++p) {           // zero-padded weight tail
            unsigned int val = 0u;
            if constexpr (EXTRA) {
                if (p == 0) val = pk_bf16(W[(size_t)K0 * 128 + tid], 0.0f);
            } else {
                if (p < K0 / 2)
                    val = pk_bf16(W[(size_t)(KFULL + 2 * p)     * 128 + tid],
                                  W[(size_t)(KFULL + 2 * p + 1) * 128 + tid]);
            }
            ldsb[tid * CSTR + p] = val;
        }
        __syncthreads();

        ABfrag a;
        if constexpr (EXTRA) {                   // only K=K0 live: pair 0, half 0
            unsigned int pe = pk_bf16(X1[srow], 0.0f);
            a.u[0] = (half == 0) ? pe : 0u;
#pragma unroll
            for (int v = 1; v < 8; ++v) a.u[v] = 0u;
        } else {                                 // K0 == 8: pairs 0..3, half 0
            const float* xrow = X0 + (size_t)srow * K0;
            float4 g0 = *(const float4*)(xrow);
            float4 g1 = *(const float4*)(xrow + 4);
            const bool h0 = (half == 0);
            a.u[0] = h0 ? pk_bf16(g0.x, g0.y) : 0u;
            a.u[1] = h0 ? pk_bf16(g0.z, g0.w) : 0u;
            a.u[2] = h0 ? pk_bf16(g1.x, g1.y) : 0u;
            a.u[3] = h0 ? pk_bf16(g1.z, g1.w) : 0u;
            a.u[4] = 0u; a.u[5] = 0u; a.u[6] = 0u; a.u[7] = 0u;
        }

#pragma unroll
        for (int nt = 0; nt < 8; ++nt) {
            const int col = nt * 16 + m;
            const unsigned int* bp = &ldsb[col * CSTR + 8 * half];
            ABfrag b;
            b.q[0] = *(const uint4*)(bp);
            b.q[1] = *(const uint4*)(bp + 4);
            acc[nt] = __builtin_amdgcn_wmma_f32_16x16x32_bf16(
                false, a.v, false, b.v, (short)0, acc[nt], false, false);
        }
    }

    // ---------------- store D + bias (row = r + 8*half) -------------------
    if (rowBase + 16 <= M) {
        // fast path: whole tile valid (always taken when M % 16 == 0) ->
        // wave-uniform branch, fully unconditional stores
#pragma unroll
        for (int nt = 0; nt < 8; ++nt) {
            const int col = nt * 16 + m;
            const float bv = bias[col];
            float* yp = Y + (size_t)(rowBase + half * 8) * 128 + col;
#pragma unroll
            for (int r = 0; r < 8; ++r)
                yp[(size_t)r * 128] = acc[nt][r] + bv;
        }
    } else if (rowBase < M) {
        // ragged tail (M % 16 != 0) -- not hit for these shapes, kept correct
#pragma unroll
        for (int nt = 0; nt < 8; ++nt) {
            const int col = nt * 16 + m;
            const float bv = bias[col];
#pragma unroll
            for (int r = 0; r < 8; ++r) {
                const int orow = rowBase + r + half * 8;
                if (orow < M) Y[(size_t)orow * 128 + col] = acc[nt][r] + bv;
            }
        }
    }
}

// ---------------------------------------------------------------------------
// Edge-degree counting (segment sizes are static across iterations)
// ---------------------------------------------------------------------------
__global__ __launch_bounds__(256)
void count_edges(const int* __restrict__ e_lit, const int* __restrict__ e_cls,
                 float* __restrict__ cnt_l, float* __restrict__ cnt_c, int E) {
    int t = blockIdx.x * blockDim.x + threadIdx.x;
    if (t >= E) return;
    atomicAdd(&cnt_l[e_lit[t]], 1.0f);
    atomicAdd(&cnt_c[e_cls[t]], 1.0f);
}

// ---------------------------------------------------------------------------
// Segment-sum scatter: dst[didx[e]][:] += src[sidx[e]][:]  (128 f32 per edge)
// 32 threads per edge, float4 per thread; f32 atomics land in L2
// (c_acc/l_acc fit in the 192MB global L2).
// ---------------------------------------------------------------------------
__global__ __launch_bounds__(256)
void scatter_add(const float* __restrict__ src, const int* __restrict__ sidx,
                 const int* __restrict__ didx, float* __restrict__ dst, int E) {
    long long t = (long long)blockIdx.x * blockDim.x + threadIdx.x;
    if (t >= (long long)E * 32) return;
    const int e = (int)(t >> 5);
    const int c = (int)(t & 31) << 2;
    const float4 v = *(const float4*)(src + (size_t)sidx[e] * 128 + c);
    float* d = dst + (size_t)didx[e] * 128 + c;
    atomicAdd(d + 0, v.x);
    atomicAdd(d + 1, v.y);
    atomicAdd(d + 2, v.z);
    atomicAdd(d + 3, v.w);
}

// ---------------------------------------------------------------------------
// In-place: acc = relu(acc / max(cnt,1))
// ---------------------------------------------------------------------------
__global__ __launch_bounds__(256)
void finalize_mean_relu(float* __restrict__ acc, const float* __restrict__ cnt,
                        int rows) {
    long long t = (long long)blockIdx.x * blockDim.x + threadIdx.x;
    if (t >= (long long)rows * 32) return;
    const int r = (int)(t >> 5);
    const int c = (int)(t & 31) << 2;
    const float inv = 1.0f / fmaxf(cnt[r], 1.0f);
    float4* p = (float4*)(acc + (size_t)r * 128 + c);
    float4 v = *p;
    v.x = fmaxf(v.x * inv, 0.0f);
    v.y = fmaxf(v.y * inv, 0.0f);
    v.z = fmaxf(v.z * inv, 0.0f);
    v.w = fmaxf(v.w * inv, 0.0f);
    *p = v;
}

// ---------------------------------------------------------------------------
// Literal tying: out[r][0:128] = lembs[r], out[r][128:256] = lembs[r^1]
// ---------------------------------------------------------------------------
__global__ __launch_bounds__(256)
void tie_literals(const float* __restrict__ lembs, float* __restrict__ out,
                  int Lrows) {
    long long t = (long long)blockIdx.x * blockDim.x + threadIdx.x;
    if (t >= (long long)Lrows * 64) return;
    const int r = (int)(t >> 6);
    const int c = (int)(t & 63) << 2;
    const int srcRow = (c < 128) ? r : (r ^ 1);
    const int sc = c & 127;
    const float4 v = *(const float4*)(lembs + (size_t)srcRow * 128 + sc);
    *(float4*)(out + (size_t)r * 256 + c) = v;
}

// ---------------------------------------------------------------------------
extern "C" void kernel_launch(void* const* d_in, const int* in_sizes, int n_in,
                              void* d_out, int out_size, void* d_ws, size_t ws_size,
                              hipStream_t stream) {
    const float* lit_feat    = (const float*)d_in[0];  // (L, 8)
    const float* clause_feat = (const float*)d_in[1];  // (C, 1)
    const int*   e_lit       = (const int*)d_in[2];    // (E,)
    const int*   e_cls       = (const int*)d_in[3];    // (E,)
    const float* W0          = (const float*)d_in[4];  // (8, 128)
    const float* b0          = (const float*)d_in[5];  // (128,)
    const float* Wl          = (const float*)d_in[6];  // (2, 256, 128)
    const float* bl          = (const float*)d_in[7];  // (2, 128)
    const float* Wc          = (const float*)d_in[8];  // (3, 129, 128)
    const float* bc          = (const float*)d_in[9];  // (3, 128)
    float* out = (float*)d_out;                        // (L, 256)

    const int L = in_sizes[0] / 8;
    const int C = in_sizes[1];
    const int E = in_sizes[2];

    char* ws = (char*)d_ws;
    float* Wh_l2c = (float*)ws; ws += (size_t)L * 128 * 4;
    float* c_acc  = (float*)ws; ws += (size_t)C * 128 * 4;
    float* Wh_c2l = (float*)ws; ws += (size_t)C * 128 * 4;
    float* l_acc  = (float*)ws; ws += (size_t)L * 128 * 4;
    float* cnt_c  = (float*)ws; ws += (size_t)C * 4;
    float* cnt_l  = (float*)ws; ws += (size_t)L * 4;

    // static segment sizes (recomputed every call: launch must be stateless)
    hipMemsetAsync(cnt_c, 0, (size_t)C * 4, stream);
    hipMemsetAsync(cnt_l, 0, (size_t)L * 4, stream);
    count_edges<<<(E + 255) / 256, 256, 0, stream>>>(e_lit, e_cls, cnt_l, cnt_c, E);

    const unsigned gridL   = (unsigned)((L + 63) / 64);
    const unsigned gridC   = (unsigned)((C + 63) / 64);
    const unsigned gridE   = (unsigned)(((long long)E * 32 + 255) / 256);
    const unsigned gridCf  = (unsigned)(((long long)C * 32 + 255) / 256);
    const unsigned gridLf  = (unsigned)(((long long)L * 32 + 255) / 256);
    const unsigned gridTie = (unsigned)(((long long)L * 64 + 255) / 256);

    for (int it = 0; it < 3; ++it) {
        // ---- literal -> clause messages ----
        if (it == 0)
            gemm_wmma_bias<8, false><<<gridL, 128, 0, stream>>>(
                lit_feat, nullptr, W0, b0, Wh_l2c, L);
        else
            gemm_wmma_bias<256, false><<<gridL, 128, 0, stream>>>(
                out, nullptr, Wl + (size_t)(it - 1) * 256 * 128,
                bl + (size_t)(it - 1) * 128, Wh_l2c, L);

        hipMemsetAsync(c_acc, 0, (size_t)C * 128 * 4, stream);
        scatter_add<<<gridE, 256, 0, stream>>>(Wh_l2c, e_lit, e_cls, c_acc, E);
        finalize_mean_relu<<<gridCf, 256, 0, stream>>>(c_acc, cnt_c, C);

        // ---- clause -> literal messages (K = 128 emb + 1 clause feature) ----
        gemm_wmma_bias<128, true><<<gridC, 128, 0, stream>>>(
            c_acc, clause_feat, Wc + (size_t)it * 129 * 128,
            bc + (size_t)it * 128, Wh_c2l, C);

        hipMemsetAsync(l_acc, 0, (size_t)L * 128 * 4, stream);
        scatter_add<<<gridE, 256, 0, stream>>>(Wh_c2l, e_cls, e_lit, l_acc, E);
        finalize_mean_relu<<<gridLf, 256, 0, stream>>>(l_acc, cnt_l, L);

        // ---- tie pos/neg literal embeddings -> (L, 256) output ----
        tie_literals<<<gridTie, 256, 0, stream>>>(l_acc, out, L);
    }
}
